// RTDetrV2MultiscaleDeformableAttention_56684978372698
// MI455X (gfx1250) — matrix operand order repaired and checked by reading, not compiled, not run on hardware
//
#include <hip/hip_runtime.h>
#include <hip/hip_bf16.h>

typedef __attribute__((ext_vector_type(16))) _Float16 v16h;
typedef __attribute__((ext_vector_type(8)))  _Float16 v8h;
typedef __attribute__((ext_vector_type(8)))  float    v8f;

#define B_  32
#define Q_  300
#define D_  256
#define H_  8
#define C_  32
#define S_  8400
#define TP_ 12

// ---------------------------------------------------------------------------
// Pack a float32 weight matrix (optionally two concatenated column blocks)
// into the per-lane f16 B-fragment layout of V_WMMA_F32_16X16X32_F16.
// Fragment storage: frag[((ntile*8 + kstep)*32 + lane)*16 + j]
//   lane: column n = ntile*16 + (lane&15); half = lane>>4
//   element j (0..15) <-> K = kstep*32 + half*16 + j   (contiguous k order, same as the A fragment)
// ---------------------------------------------------------------------------
__global__ void pack_wfrag(const float* __restrict__ W0, const float* __restrict__ W1,
                           int split, int ld0, int ld1, int ntiles,
                           _Float16* __restrict__ dst)
{
    const int t = blockIdx.x * blockDim.x + threadIdx.x;
    const int total = ntiles * 8 * 32;
    if (t >= total) return;
    const int lane  = t & 31;
    const int ks    = (t >> 5) & 7;
    const int ntile = t >> 8;
    const int lrow  = lane & 15;
    const int half  = lane >> 4;
    const int n     = ntile * 16 + lrow;
    _Float16* out = dst + (size_t)t * 16;
#pragma unroll
    for (int j = 0; j < 16; ++j) {
        const int k = ks * 32 + half * 16 + j;
        const float v = (n < split) ? W0[(size_t)k * ld0 + n]
                                    : W1[(size_t)k * ld1 + (n - split)];
        out[j] = (_Float16)v;
    }
}

// ---------------------------------------------------------------------------
// WMMA GEMM:  C[M,N] = A[M,256] * Wfrag + bias,  f16 multiply / f32 accumulate.
// One wave computes a 16 x (TN*16) tile.  blockDim = 128 (4 waves).
// grid.x = Mtiles/4, grid.y = N/(TN*16).
// F16OUT selects f16 vs f32 output storage.
// Epilogue transposes through LDS for coalesced vector stores.
// ---------------------------------------------------------------------------
template <int TN, bool F16OUT>
__global__ __launch_bounds__(128)
void gemm_wmma(const float* __restrict__ A, const _Float16* __restrict__ Bfrag,
               const float* __restrict__ bias0, const float* __restrict__ bias1,
               int split, int M, int N, void* __restrict__ Cout_)
{
    __shared__ __align__(16) float sh[4][256];
    const int lane  = threadIdx.x & 31;
    const int wave  = threadIdx.x >> 5;
    const int mtile = blockIdx.x * 4 + wave;
    const int half  = lane >> 4;
    const int lrow  = lane & 15;
    const int row   = mtile * 16 + lrow;
    if (row >= M) return;                       // uniform per wave (M % 16 == 0)
    const float* __restrict__ Arow = A + (size_t)row * 256;
    const int ntile0 = blockIdx.y * TN;

    v8f acc[TN];
#pragma unroll
    for (int t = 0; t < TN; ++t) acc[t] = (v8f){0.f,0.f,0.f,0.f,0.f,0.f,0.f,0.f};

    for (int ks = 0; ks < 8; ++ks) {            // K = 256 = 8 * 32
        // A fragment (16x32 f16): lane row = lrow;
        // VGPR0-3 hold K = half*16 .. +7, VGPR4-7 hold K = half*16+8 .. +7
        const int k0 = ks * 32 + half * 16;
        v16h a;
#pragma unroll
        for (int j = 0; j < 8; ++j) {
            a[j]     = (_Float16)Arow[k0 + j];
            a[8 + j] = (_Float16)Arow[k0 + 8 + j];
        }
#pragma unroll
        for (int t = 0; t < TN; ++t) {
            const v16h b = *(const v16h*)(Bfrag +
                (((size_t)(ntile0 + t) * 8 + ks) * 32 + lane) * 16);
            acc[t] = __builtin_amdgcn_wmma_f32_16x16x32_f16(
                false, a, false, b, (short)0, acc[t], false, false);
        }
    }

    // Epilogue: bias + LDS transpose + coalesced store.
    float* s = sh[wave];
#pragma unroll
    for (int t = 0; t < TN; ++t) {
        const int n  = (ntile0 + t) * 16 + lrow;
        const float bv = (n < split) ? bias0[n] : bias1[n - split];
        // D layout: lane column = lrow, VGPR i -> M = half*8 + i
#pragma unroll
        for (int i = 0; i < 8; ++i)
            s[(half * 8 + i) * 16 + lrow] = acc[t][i] + bv;
        // same-wave LDS RAW; DS ops are in-order per wave
        const int r     = lane >> 1;            // output row within tile
        const int cbase = (lane & 1) * 8;       // 8 cols per lane
        const size_t m  = (size_t)mtile * 16 + r;
        const int n0    = (ntile0 + t) * 16;
        const float4 v0 = *(const float4*)&s[r * 16 + cbase];
        const float4 v1 = *(const float4*)&s[r * 16 + cbase + 4];
        if (F16OUT) {
            v8h hv;
            hv[0] = (_Float16)v0.x; hv[1] = (_Float16)v0.y;
            hv[2] = (_Float16)v0.z; hv[3] = (_Float16)v0.w;
            hv[4] = (_Float16)v1.x; hv[5] = (_Float16)v1.y;
            hv[6] = (_Float16)v1.z; hv[7] = (_Float16)v1.w;
            *(v8h*)((_Float16*)Cout_ + m * (size_t)N + n0 + cbase) = hv;
        } else {
            float* Cout = (float*)Cout_;
            *(float4*)&Cout[m * (size_t)N + n0 + cbase]     = v0;
            *(float4*)&Cout[m * (size_t)N + n0 + cbase + 4] = v1;
        }
    }
}

// ---------------------------------------------------------------------------
// Deformable sampling: one wave per (b, q, h); lane = channel c.
// proj layout per (b,q) row of 288: [0..191] offsets (h*24 + p*2 + xy),
// [192..287] attention logits (192 + h*12 + p).
// gx = loc_x * W - 0.5 (algebraic simplification of the grid_sample mapping).
// value is stored f16 (intermediate only; consumed in convex combinations).
// ---------------------------------------------------------------------------
__global__ __launch_bounds__(256)
void msda_sample(const _Float16* __restrict__ value,  // [B*S, 256] f16
                 const float* __restrict__ proj,      // [B*Q, 288]
                 const float* __restrict__ refp,      // [B*Q, 4]
                 float* __restrict__ outattn)         // [B*Q, 256]
{
    const int lane = threadIdx.x & 31;
    const int wid  = blockIdx.x * (blockDim.x >> 5) + (threadIdx.x >> 5);
    if (wid >= B_ * Q_ * H_) return;
    const int h  = wid & (H_ - 1);
    const int bq = wid >> 3;
    const int b  = bq / Q_;

    const float* __restrict__ pr = proj + (size_t)bq * 288;
    const float r0 = refp[bq * 4 + 0];
    const float r1 = refp[bq * 4 + 1];
    const float sx = refp[bq * 4 + 2] * 0.125f;   // (1/N_POINTS) * OFFSET_SCALE
    const float sy = refp[bq * 4 + 3] * 0.125f;

    // softmax over 12 logits (redundant per lane; cheap VALU)
    float lg[TP_];
    float mx = -3.0e38f;
#pragma unroll
    for (int p = 0; p < TP_; ++p) { lg[p] = pr[192 + h * TP_ + p]; mx = fmaxf(mx, lg[p]); }
    float sum = 0.f;
#pragma unroll
    for (int p = 0; p < TP_; ++p) { lg[p] = __expf(lg[p] - mx); sum += lg[p]; }
    const float inv = 1.0f / sum;

    const int   lvlW[3] = {80, 40, 20};
    const int   lvlH[3] = {80, 40, 20};
    const int   lvlS[3] = {0, 6400, 8000};
    const float* __restrict__ offp = pr + h * (TP_ * 2);

    float acc = 0.f;
#pragma unroll
    for (int p = 0; p < TP_; ++p) {
        const int lvl = p >> 2;
        const int Wl = lvlW[lvl], Hl = lvlH[lvl], st = lvlS[lvl];
        const float aw = lg[p] * inv;
        const float lx = r0 + offp[p * 2 + 0] * sx;
        const float ly = r1 + offp[p * 2 + 1] * sy;
        const float gx = lx * (float)Wl - 0.5f;
        const float gy = ly * (float)Hl - 0.5f;
        const float x0f = floorf(gx), y0f = floorf(gy);
        const int   x0 = (int)x0f,   y0 = (int)y0f;
        const float wx1 = gx - x0f, wx0 = 1.f - wx1;
        const float wy1 = gy - y0f, wy0 = 1.f - wy1;
        const size_t base = ((size_t)b * S_ + st) * 256 + h * C_ + lane;
#pragma unroll
        for (int ty = 0; ty < 2; ++ty) {
            const int iy = y0 + ty;
            if (iy < 0 || iy >= Hl) continue;
            const float wy = ty ? wy1 : wy0;
#pragma unroll
            for (int tx = 0; tx < 2; ++tx) {
                const int ix = x0 + tx;
                if (ix < 0 || ix >= Wl) continue;
                const float w = aw * wy * (tx ? wx1 : wx0);
                acc += w * (float)value[base + ((size_t)iy * Wl + ix) * 256];
            }
        }
    }
    outattn[(size_t)bq * 256 + h * C_ + lane] = acc;
}

// ---------------------------------------------------------------------------
extern "C" void kernel_launch(void* const* d_in, const int* in_sizes, int n_in,
                              void* d_out, int out_size, void* d_ws, size_t ws_size,
                              hipStream_t stream)
{
    (void)in_sizes; (void)n_in; (void)out_size; (void)ws_size;

    const float* hidden = (const float*)d_in[0];   // [B,Q,256]
    const float* enc    = (const float*)d_in[1];   // [B,S,256]
    const float* refp   = (const float*)d_in[2];   // [B,Q,1,4]
    const float* W_off  = (const float*)d_in[3];   // [256,192]
    const float* b_off  = (const float*)d_in[4];
    const float* W_attn = (const float*)d_in[5];   // [256,96]
    const float* b_attn = (const float*)d_in[6];
    const float* W_val  = (const float*)d_in[7];   // [256,256]
    const float* b_val  = (const float*)d_in[8];
    const float* W_out  = (const float*)d_in[9];   // [256,256]
    const float* b_out  = (const float*)d_in[10];
    float* out = (float*)d_out;

    // workspace layout (all offsets 32B aligned)
    char* ws = (char*)d_ws;
    const size_t VAL_BYTES   = (size_t)B_ * S_ * 256 * 2;       // 137,625,600 (f16)
    const size_t PROJ_BYTES  = (size_t)B_ * Q_ * 288 * 4;       //  11,059,200
    const size_t OATTN_BYTES = (size_t)B_ * Q_ * 256 * 4;       //   9,830,400
    _Float16*  value    = (_Float16*)(ws);
    float*     proj     = (float*)(ws + VAL_BYTES);
    float*     outattn  = (float*)(ws + VAL_BYTES + PROJ_BYTES);
    _Float16*  frag_val = (_Float16*)(ws + VAL_BYTES + PROJ_BYTES + OATTN_BYTES);
    _Float16*  frag_oa  = frag_val + (size_t)16 * 8 * 32 * 16;  // 16 ntiles
    _Float16*  frag_out = frag_oa  + (size_t)18 * 8 * 32 * 16;  // 18 ntiles

    // 1) pack weights into WMMA B fragments (f16)
    pack_wfrag<<<(16 * 256 + 255) / 256, 256, 0, stream>>>(W_val, W_val, 256, 256, 256, 16, frag_val);
    pack_wfrag<<<(18 * 256 + 255) / 256, 256, 0, stream>>>(W_off, W_attn, 192, 192, 96, 18, frag_oa);
    pack_wfrag<<<(16 * 256 + 255) / 256, 256, 0, stream>>>(W_out, W_out, 256, 256, 256, 16, frag_out);

    // 2) value projection: [268800,256] x [256,256] -> f16, A read exactly once
    gemm_wmma<16, true><<<dim3((B_ * S_ / 16) / 4, 1), 128, 0, stream>>>(
        enc, frag_val, b_val, b_val, 256, B_ * S_, 256, (void*)value);

    // 3) fused offset+attention projection: [9600,256] x [256,288]
    gemm_wmma<3, false><<<dim3((B_ * Q_ / 16) / 4, 6), 128, 0, stream>>>(
        hidden, frag_oa, b_off, b_attn, 192, B_ * Q_, 288, (void*)proj);

    // 4) softmax + bilinear sampling + weighted sum
    msda_sample<<<(B_ * Q_ * H_) / 8, 256, 0, stream>>>(value, proj, refp, outattn);

    // 5) output projection: [9600,256] x [256,256] -> d_out
    gemm_wmma<4, false><<<dim3((B_ * Q_ / 16) / 4, 4), 128, 0, stream>>>(
        outattn, frag_out, b_out, b_out, 256, B_ * Q_, 256, (void*)out);
}
